// Gemma4VisionPatchEmbedder_33646773797304
// MI455X (gfx1250) — matrix operand order repaired and checked by reading, not compiled, not run on hardware
//
#include <hip/hip_runtime.h>

// Problem constants (from reference): B=8, N=4096, P=256, H=1152, D=588
namespace {
constexpr int Bb = 8, Nn = 4096, Pp = 256, Hh = 1152, Dd = 588;
constexpr int M_TOTAL = Bb * Nn;          // 32768 rows
constexpr int BM = 128, BN = 128, BK = 32;
constexpr int KP = 608;                   // K padded to 19*32 (zero-filled)
constexpr int KT = KP / BK;               // 19 k-steps, no guards
constexpr int GRP = KP / 4;               // 152 float4-groups per padded row
}

typedef __attribute__((ext_vector_type(16))) __bf16 v16bf;
typedef __attribute__((ext_vector_type(8)))  __bf16 v8bf;
typedef __attribute__((ext_vector_type(4)))  __bf16 v4bf;
typedef __attribute__((ext_vector_type(8)))  float  v8f;
typedef __attribute__((ext_vector_type(4)))  float  v4f;
typedef __attribute__((ext_vector_type(4)))  unsigned int v4u;
typedef __attribute__((ext_vector_type(8)))  int    v8i;
typedef __attribute__((ext_vector_type(4)))  int    v4i;

// ---------------------------------------------------------------------------
// Pass 1: fp32 -> bf16 conversion with affine fold (A) and zero K-padding.
//   A_bf16 [32768, 608] = 2*pv - 1      (cols 588..607 = 0)
//   B_bf16 [ 1152, 608] = W             (cols 588..607 = 0)
// ---------------------------------------------------------------------------
__global__ __launch_bounds__(256)
void convert_bf16(const float* __restrict__ pv, const float* __restrict__ W,
                  __bf16* __restrict__ Abf, __bf16* __restrict__ Bbf) {
  const long g = (long)blockIdx.x * 256 + threadIdx.x;
  const long groupsA = (long)M_TOTAL * GRP;
  const long groupsB = (long)Hh * GRP;
  if (g < groupsA) {
    const int r = (int)(g / GRP);
    const int c = (int)(g % GRP) * 4;
    v4bf o = {};
    if (c < Dd) {
      v4f x = *(const v4f*)(pv + (size_t)r * Dd + c);
#pragma unroll
      for (int j = 0; j < 4; ++j) o[j] = (__bf16)(2.f * x[j] - 1.f);
    }
    *(v4bf*)(Abf + (size_t)r * KP + c) = o;
  } else if (g < groupsA + groupsB) {
    const long gb = g - groupsA;
    const int r = (int)(gb / GRP);
    const int c = (int)(gb % GRP) * 4;
    v4bf o = {};
    if (c < Dd) {
      v4f x = *(const v4f*)(W + (size_t)r * Dd + c);
#pragma unroll
      for (int j = 0; j < 4; ++j) o[j] = (__bf16)x[j];
    }
    *(v4bf*)(Bbf + (size_t)r * KP + c) = o;
  }
}

// ---------------------------------------------------------------------------
// TDM descriptor build + issue: 2-D tile (32 cols x 128 rows, bf16) from a
// row-major [rows, KP] bf16 tensor into LDS (compact 128x32 row-major tile).
// D# packing per CDNA5 ISA 8.3/8.4: group0 {count=1, lds_addr, global_addr,
// type=2}; group1 {data_size=2B, tensor_dim0=KP, tensor_dim1=rows,
// tile_dim0=32, tile_dim1=128, tensor_dim0_stride=KP}; groups 2/3 zero (2-D).
// This toolchain's builtin takes the 6-arg form with a trailing zero int32x8.
// ---------------------------------------------------------------------------
__device__ __forceinline__ void tdm_load_tile(unsigned int lds_byte_off,
                                              const __bf16* gsrc,
                                              unsigned int rows) {
  const unsigned long long ga = (unsigned long long)(uintptr_t)gsrc;
  v4u g0;
  g0[0] = 1u;                                             // count=1, user mode
  g0[1] = lds_byte_off;                                   // lds_addr
  g0[2] = (unsigned int)ga;                               // global_addr[31:0]
  g0[3] = (unsigned int)((ga >> 32) & 0x1FFFFFFull) | (2u << 30); // [56:32]+type=2
  v8i g1;
  g1[0] = (int)(1u << 16);                                // data_size=1 (2 bytes)
  g1[1] = (int)((KP & 0xFFFF) << 16);                     // tensor_dim0 lo16
  g1[2] = (int)(((KP >> 16) & 0xFFFF) | ((rows & 0xFFFF) << 16)); // dim0 hi | dim1 lo
  g1[3] = (int)(((rows >> 16) & 0xFFFF) | (32u << 16));   // dim1 hi | tile_dim0=32
  g1[4] = (int)128;                                       // tile_dim1=128, tile_dim2=0
  g1[5] = (int)KP;                                        // tensor_dim0_stride lo32
  g1[6] = 0;                                              // stride hi | dim1_stride lo
  g1[7] = 0;                                              // dim1_stride (unused, 2-D)
  v4i z4 = {0, 0, 0, 0};                                  // groups 2/3: dims 3/4 unused
  v8i z8 = {0, 0, 0, 0, 0, 0, 0, 0};                      // trailing arg: zero-filled
  __builtin_amdgcn_tensor_load_to_lds(g0, g1, z4, z4, z8, 0);
}

// ---------------------------------------------------------------------------
// Pass 2: bf16 WMMA GEMM with TDM double-buffered LDS tiles + fused PE epilogue.
// ---------------------------------------------------------------------------
__global__ __launch_bounds__(256)
void patch_embed_gemm_tdm(const __bf16* __restrict__ Abf,  // [32768, 608]
                          const __bf16* __restrict__ Bbf,  // [1152, 608]
                          const int*    __restrict__ pos,  // [32768, 2]
                          const int*    __restrict__ pad,  // [32768]
                          const float*  __restrict__ table,// [2, 256, 1152]
                          float*        __restrict__ out) {// [32768, 1152]
  __shared__ __bf16 sA[2][BM][BK];   // 16 KB double-buffered A tile
  __shared__ __bf16 sB[2][BN][BK];   // 16 KB double-buffered W tile
  __shared__ int   sPos0[BM];
  __shared__ int   sPos1[BM];
  __shared__ float sMask[BM];        // 1.0 = add PE, 0.0 = padded (branchless)

  const int tid    = threadIdx.x;
  const int lane   = tid & 31;
  const int wid    = tid >> 5;
  const int waveM  = (wid & 3) * 32;   // 4 waves along M, 32 rows each
  const int waveN  = (wid >> 2) * 64;  // 2 waves along N, 64 cols each
  const int blockN = blockIdx.x * BN;
  const int blockM = blockIdx.y * BM;

  if (tid < BM) {
    const int row = blockM + tid;
    int p0 = pos[row * 2 + 0];
    int p1 = pos[row * 2 + 1];
    sPos0[tid] = p0 < 0 ? 0 : p0;
    sPos1[tid] = p1 < 0 ? 0 : p1;
    sMask[tid] = pad[row] ? 0.f : 1.f;
  }

  // Generic LDS pointers truncate to the LDS byte offset (flat-aperture rule).
  const unsigned int ldsA[2] = {(unsigned int)(uintptr_t)&sA[0][0][0],
                                (unsigned int)(uintptr_t)&sA[1][0][0]};
  const unsigned int ldsB[2] = {(unsigned int)(uintptr_t)&sB[0][0][0],
                                (unsigned int)(uintptr_t)&sB[1][0][0]};
  const __bf16* aBase = Abf + (size_t)blockM * KP;
  const __bf16* bBase = Bbf + (size_t)blockN * KP;

  // A fragment: lanes 0-15 hold K in {0..7,16..23}, lanes 16-31 {8..15,24..31}.
  const int halfK = (lane >> 4) * 8;
  // B fragment: lanes 0-15 hold K in {0..7,16..23}, lanes 16-31 {8..15,24..31}.
  const int koffB = (lane >> 4) * 8;
  const int rsel  = lane & 15;
  auto lfragA = [&](const __bf16* rowBase) -> v16bf {
    v8bf lo = *(const v8bf*)(rowBase + halfK);
    v8bf hi = *(const v8bf*)(rowBase + halfK + 16);
    return __builtin_shufflevector(lo, hi, 0, 1, 2, 3, 4, 5, 6, 7,
                                           8, 9, 10, 11, 12, 13, 14, 15);
  };
  auto lfragB = [&](const __bf16* rowBase) -> v16bf {
    v8bf lo = *(const v8bf*)(rowBase + koffB);
    v8bf hi = *(const v8bf*)(rowBase + koffB + 16);
    return __builtin_shufflevector(lo, hi, 0, 1, 2, 3, 4, 5, 6, 7,
                                           8, 9, 10, 11, 12, 13, 14, 15);
  };

  v8f acc[2][4];
#pragma unroll
  for (int t = 0; t < 2; ++t)
#pragma unroll
    for (int u = 0; u < 4; ++u)
      acc[t][u] = (v8f){0.f, 0.f, 0.f, 0.f, 0.f, 0.f, 0.f, 0.f};

  // Prologue: wave 0 DMAs tile 0 into buffer 0.
  if (wid == 0) {
    tdm_load_tile(ldsA[0], aBase, (unsigned int)M_TOTAL);
    tdm_load_tile(ldsB[0], bBase, (unsigned int)Hh);
    __builtin_amdgcn_s_wait_tensorcnt(0);
  }
  __syncthreads();

  for (int kt = 0; kt < KT; ++kt) {
    const int cur = kt & 1;
    // Kick off next tile's DMA before compute so TDM overlaps the WMMAs.
    if (wid == 0 && kt + 1 < KT) {
      const int nxt = (kt + 1) & 1;
      tdm_load_tile(ldsA[nxt], aBase + (kt + 1) * BK, (unsigned int)M_TOTAL);
      tdm_load_tile(ldsB[nxt], bBase + (kt + 1) * BK, (unsigned int)Hh);
    }

    v16bf af[2], bf[4];
#pragma unroll
    for (int t = 0; t < 2; ++t)
      af[t] = lfragA(&sA[cur][waveM + t * 16 + rsel][0]);
#pragma unroll
    for (int u = 0; u < 4; ++u)
      bf[u] = lfragB(&sB[cur][waveN + u * 16 + rsel][0]);

#pragma unroll
    for (int t = 0; t < 2; ++t)
#pragma unroll
      for (int u = 0; u < 4; ++u)
        acc[t][u] = __builtin_amdgcn_wmma_f32_16x16x32_bf16(
            false, af[t], false, bf[u], (short)0, acc[t][u], false, false);

    if (wid == 0 && kt + 1 < KT) __builtin_amdgcn_s_wait_tensorcnt(0);
    __syncthreads();
  }

  // Fused epilogue: branchless position-embedding gather + store.
  const float* __restrict__ t0 = table;
  const float* __restrict__ t1 = table + (size_t)Pp * Hh;
  const int rowHalf = (lane >> 4) * 8;
#pragma unroll
  for (int t = 0; t < 2; ++t) {
#pragma unroll
    for (int r = 0; r < 8; ++r) {
      const int rowL = waveM + t * 16 + rowHalf + r;
      const float m  = sMask[rowL];
      const size_t o0 = (size_t)sPos0[rowL] * Hh;
      const size_t o1 = (size_t)sPos1[rowL] * Hh;
      const size_t ob = (size_t)(blockM + rowL) * Hh;
#pragma unroll
      for (int u = 0; u < 4; ++u) {
        const int h = blockN + waveN + u * 16 + rsel;
        const float pe = t0[o0 + h] + t1[o1 + h];
        out[ob + h] = fmaf(m, pe, acc[t][u][r]);
      }
    }
  }
}

// ---------------------------------------------------------------------------
// Fallback: single-pass fused kernel (fp32 global loads, in-loop conversion).
// Used only when ws_size can't hold the bf16 staging buffers.
// ---------------------------------------------------------------------------
__global__ __launch_bounds__(256)
void patch_embed_gemm_fused(const float* __restrict__ pv,
                            const int*   __restrict__ pos,
                            const int*   __restrict__ pad,
                            const float* __restrict__ W,
                            const float* __restrict__ table,
                            float*       __restrict__ out) {
  __shared__ __bf16 sA[2][BM][BK];
  __shared__ __bf16 sB[2][BN][BK];
  __shared__ int   sPos0[BM];
  __shared__ int   sPos1[BM];
  __shared__ float sMask[BM];

  const int tid    = threadIdx.x;
  const int lane   = tid & 31;
  const int wid    = tid >> 5;
  const int waveM  = (wid & 3) * 32;
  const int waveN  = (wid >> 2) * 64;
  const int blockN = blockIdx.x * BN;
  const int blockM = blockIdx.y * BM;

  if (tid < BM) {
    const int row = blockM + tid;
    int p0 = pos[row * 2 + 0];
    int p1 = pos[row * 2 + 1];
    sPos0[tid] = p0 < 0 ? 0 : p0;
    sPos1[tid] = p1 < 0 ? 0 : p1;
    sMask[tid] = pad[row] ? 0.f : 1.f;
  }

  v4f ra[4], rb[4];
  auto gload = [&](int kOff) {
#pragma unroll
    for (int i = 0; i < 4; ++i) {
      const int c = tid + 256 * i;
      const int r = c >> 3;
      const int k = kOff + (c & 7) * 4;
      v4f za = {0.f, 0.f, 0.f, 0.f};
      v4f zb = {0.f, 0.f, 0.f, 0.f};
      if (k < Dd) {
        za = *(const v4f*)(pv + (size_t)(blockM + r) * Dd + k);
        zb = *(const v4f*)(W  + (size_t)(blockN + r) * Dd + k);
      }
      ra[i] = za;
      rb[i] = zb;
    }
  };
  auto sstore = [&](int buf) {
#pragma unroll
    for (int i = 0; i < 4; ++i) {
      const int c  = tid + 256 * i;
      const int r  = c >> 3;
      const int kc = (c & 7) * 4;
      v4bf pa, pb;
#pragma unroll
      for (int j = 0; j < 4; ++j) {
        pa[j] = (__bf16)(2.f * ra[i][j] - 1.f);
        pb[j] = (__bf16)rb[i][j];
      }
      *(v4bf*)&sA[buf][r][kc] = pa;
      *(v4bf*)&sB[buf][r][kc] = pb;
    }
  };

  const int halfK = (lane >> 4) * 8;
  const int koffB = (lane >> 4) * 8;
  const int rsel  = lane & 15;
  auto lfragA = [&](const __bf16* rowBase) -> v16bf {
    v8bf lo = *(const v8bf*)(rowBase + halfK);
    v8bf hi = *(const v8bf*)(rowBase + halfK + 16);
    return __builtin_shufflevector(lo, hi, 0, 1, 2, 3, 4, 5, 6, 7,
                                           8, 9, 10, 11, 12, 13, 14, 15);
  };
  auto lfragB = [&](const __bf16* rowBase) -> v16bf {
    v8bf lo = *(const v8bf*)(rowBase + koffB);
    v8bf hi = *(const v8bf*)(rowBase + koffB + 16);
    return __builtin_shufflevector(lo, hi, 0, 1, 2, 3, 4, 5, 6, 7,
                                           8, 9, 10, 11, 12, 13, 14, 15);
  };

  v8f acc[2][4];
#pragma unroll
  for (int t = 0; t < 2; ++t)
#pragma unroll
    for (int u = 0; u < 4; ++u)
      acc[t][u] = (v8f){0.f, 0.f, 0.f, 0.f, 0.f, 0.f, 0.f, 0.f};

  gload(0);
  sstore(0);
  __syncthreads();

  constexpr int KT_F = (Dd + BK - 1) / BK;
  for (int kt = 0; kt < KT_F; ++kt) {
    const int cur = kt & 1;
    if (kt + 1 < KT_F) gload((kt + 1) * BK);

    v16bf af[2], bf[4];
#pragma unroll
    for (int t = 0; t < 2; ++t)
      af[t] = lfragA(&sA[cur][waveM + t * 16 + rsel][0]);
#pragma unroll
    for (int u = 0; u < 4; ++u)
      bf[u] = lfragB(&sB[cur][waveN + u * 16 + rsel][0]);

#pragma unroll
    for (int t = 0; t < 2; ++t)
#pragma unroll
      for (int u = 0; u < 4; ++u)
        acc[t][u] = __builtin_amdgcn_wmma_f32_16x16x32_bf16(
            false, af[t], false, bf[u], (short)0, acc[t][u], false, false);

    if (kt + 1 < KT_F) sstore((kt + 1) & 1);
    __syncthreads();
  }

  const float* __restrict__ t0 = table;
  const float* __restrict__ t1 = table + (size_t)Pp * Hh;
  const int rowHalf = (lane >> 4) * 8;
#pragma unroll
  for (int t = 0; t < 2; ++t) {
#pragma unroll
    for (int r = 0; r < 8; ++r) {
      const int rowL = waveM + t * 16 + rowHalf + r;
      const float m  = sMask[rowL];
      const size_t o0 = (size_t)sPos0[rowL] * Hh;
      const size_t o1 = (size_t)sPos1[rowL] * Hh;
      const size_t ob = (size_t)(blockM + rowL) * Hh;
#pragma unroll
      for (int u = 0; u < 4; ++u) {
        const int h = blockN + waveN + u * 16 + rsel;
        const float pe = t0[o0 + h] + t1[o1 + h];
        out[ob + h] = fmaf(m, pe, acc[t][u][r]);
      }
    }
  }
}

extern "C" void kernel_launch(void* const* d_in, const int* in_sizes, int n_in,
                              void* d_out, int out_size, void* d_ws, size_t ws_size,
                              hipStream_t stream) {
  const float* pv  = (const float*)d_in[0];   // pixel_values   [8,4096,588] f32
  const int*   pos = (const int*)  d_in[1];   // position ids   [8,4096,2]
  const int*   pad = (const int*)  d_in[2];   // padding mask   [8,4096]
  const float* W   = (const float*)d_in[3];   // projection     [1152,588] f32
  const float* tab = (const float*)d_in[4];   // pos-emb table  [2,256,1152] f32
  float* out = (float*)d_out;                 // [8,4096,1152] f32

  const size_t need = (size_t)(M_TOTAL + Hh) * KP * sizeof(__bf16);  // ~41.2 MB
  dim3 grid(Hh / BN, M_TOTAL / BM);           // (9, 256)

  if (ws_size >= need) {
    __bf16* Abf = (__bf16*)d_ws;
    __bf16* Bbf = Abf + (size_t)M_TOTAL * KP;
    const long groups = (long)(M_TOTAL + Hh) * GRP;
    const int cblocks = (int)((groups + 255) / 256);
    convert_bf16<<<cblocks, 256, 0, stream>>>(pv, W, Abf, Bbf);
    patch_embed_gemm_tdm<<<grid, dim3(256), 0, stream>>>(Abf, Bbf, pos, pad, tab, out);
  } else {
    patch_embed_gemm_fused<<<grid, dim3(256), 0, stream>>>(pv, pos, pad, W, tab, out);
  }
}
